// FusionTokenBlock_12884901888496
// MI455X (gfx1250) — matrix operand order repaired and checked by reading, not compiled, not run on hardware
//
#include <hip/hip_runtime.h>
#include <cstdint>
#include <cstddef>
#include <type_traits>

// ---------------------------------------------------------------------------
// CDNA5 / gfx1250 fused implementation of the dual cross-SRA block.
// All GEMM-shaped work (conv-as-GEMM, q/k/v/proj, attention einsums, FFN)
// runs through v_wmma_f32_16x16x32_f16 (f16 operands, f32 accumulate).
// ---------------------------------------------------------------------------

typedef __attribute__((ext_vector_type(16))) _Float16 v16h;
typedef __attribute__((ext_vector_type(8)))  float    v8f;

union FragH { v16h v; _Float16 h[16]; };
union AccF  { v8f  v; float     f[8];  };
union Pack8 { uint4 u; _Float16 h[8];  };

#define DIM_   512
#define HEADS_ 8
#define DH_    64
#define BATCH_ 8
#define NTOK_  4096
#define MKV_   64
#define HID_   1024

// ---------------------------------------------------------------------------
// Bilinear-resize attention bias: [B,1,64,64] -> [B,64] (8x8, half-pixel
// centers => average of the 2x2 block at rows/cols 8t+3, 8t+4), times gamma.
// ---------------------------------------------------------------------------
__global__ __launch_bounds__(256) void bias_resize_kernel(
    const float* __restrict__ bias_in, const float* __restrict__ gamma,
    float* __restrict__ out) {
  int tid = blockIdx.x * 256 + threadIdx.x;
  if (tid >= BATCH_ * MKV_) return;
  int b = tid >> 6, m = tid & 63;
  int ty = m >> 3, tx = m & 7;
  const float* p = bias_in + (size_t)b * 4096;
  int y0 = ty * 8 + 3, x0 = tx * 8 + 3;
  float v = 0.25f * (p[y0 * 64 + x0] + p[y0 * 64 + x0 + 1] +
                     p[(y0 + 1) * 64 + x0] + p[(y0 + 1) * 64 + x0 + 1]);
  out[tid] = gamma[0] * v;
}

// ---------------------------------------------------------------------------
// Generic WMMA GEMM: out = f( A[M,K] @ Bw[K,N] + bias[N] )
//   MODE 0: out = val                      (OT = f16 or f32)
//   MODE 1: out += sigmoid(*gate) * val    (OT = f32)
//   MODE 2: out = resid + val              (OT = f32)
// Block: 256 thr (8 waves), tile 128(M) x 128(N); wave tile 32x64; K-step 32.
// LDS staged with packed ds_store_b128; fragments read as ds_load_b128.
// ---------------------------------------------------------------------------
template <typename AT, typename OT, bool GELU, int MODE>
__global__ __launch_bounds__(256) void gemm_kernel(
    const AT* __restrict__ A, const float* __restrict__ Bw,
    const float* __restrict__ bias, const float* __restrict__ gate,
    const float* __restrict__ resid, OT* __restrict__ out,
    int M, int N, int K) {
  __shared__ _Float16 As[128 * 32];  // [row][kc]
  __shared__ _Float16 Bs[128 * 32];  // [n][kc]  (K pairs contiguous)

  const int tid  = threadIdx.x;
  const int lane = tid & 31, wv = tid >> 5;
  const int m0 = blockIdx.y * 128, n0 = blockIdx.x * 128;
  const int mo = (wv >> 1) * 32, no = (wv & 1) * 64;
  const int Ml = lane & 15, half = lane >> 4;
  const int hi8 = half * 8, hi16 = half * 16;

  AccF acc[2][4];
#pragma unroll
  for (int g = 0; g < 2; ++g)
#pragma unroll
    for (int nt = 0; nt < 4; ++nt)
#pragma unroll
      for (int e = 0; e < 8; ++e) acc[g][nt].f[e] = 0.0f;

  // staging coordinates
  const int arow = tid >> 1, akc = (tid & 1) * 16;   // A: 128 rows x 32 kc
  const int bn   = tid & 127, bkc = (tid >> 7) * 16; // B: 128 n x 32 kc

  for (int k0 = 0; k0 < K; k0 += 32) {
    // ---- stage A tile 128x32 -> f16, packed b128 stores
    {
      size_t off = (size_t)(m0 + arow) * K + k0 + akc;
      if constexpr (std::is_same<AT, _Float16>::value) {
        const uint4* ap = reinterpret_cast<const uint4*>(A + off);
        *reinterpret_cast<uint4*>(&As[arow * 32 + akc])     = ap[0];
        *reinterpret_cast<uint4*>(&As[arow * 32 + akc + 8]) = ap[1];
      } else {
        const float4* ap = reinterpret_cast<const float4*>(A + off);
        Pack8 p0, p1;
#pragma unroll
        for (int u = 0; u < 2; ++u) {
          float4 t0 = ap[2 * u], t1 = ap[2 * u + 1];
          Pack8& pk = u ? p1 : p0;
          pk.h[0] = (_Float16)t0.x; pk.h[1] = (_Float16)t0.y;
          pk.h[2] = (_Float16)t0.z; pk.h[3] = (_Float16)t0.w;
          pk.h[4] = (_Float16)t1.x; pk.h[5] = (_Float16)t1.y;
          pk.h[6] = (_Float16)t1.z; pk.h[7] = (_Float16)t1.w;
        }
        *reinterpret_cast<uint4*>(&As[arow * 32 + akc])     = p0.u;
        *reinterpret_cast<uint4*>(&As[arow * 32 + akc + 8]) = p1.u;
      }
    }
    // ---- stage B tile 32x128 stored [n][kc]; lane-coalesced global reads
    {
      Pack8 p0, p1;
#pragma unroll
      for (int i = 0; i < 8; ++i)
        p0.h[i] = (_Float16)Bw[(size_t)(k0 + bkc + i) * N + n0 + bn];
#pragma unroll
      for (int i = 0; i < 8; ++i)
        p1.h[i] = (_Float16)Bw[(size_t)(k0 + bkc + 8 + i) * N + n0 + bn];
      *reinterpret_cast<uint4*>(&Bs[bn * 32 + bkc])     = p0.u;
      *reinterpret_cast<uint4*>(&Bs[bn * 32 + bkc + 8]) = p1.u;
    }
    __syncthreads();

    FragH a[2];
#pragma unroll
    for (int g = 0; g < 2; ++g)
#pragma unroll
      for (int v = 0; v < 8; ++v) {
        int k = ((v & 3) << 1) + ((v >> 2) << 4) + hi8;
        a[g].h[2 * v]     = As[(mo + g * 16 + Ml) * 32 + k];
        a[g].h[2 * v + 1] = As[(mo + g * 16 + Ml) * 32 + k + 1];
      }
#pragma unroll
    for (int nt = 0; nt < 4; ++nt) {
      FragH bf;
      int Nl = no + nt * 16 + Ml;
#pragma unroll
      for (int v = 0; v < 8; ++v) {
        int kk = ((v & 3) << 1) + ((v >> 2) << 4) + hi8;
        bf.h[2 * v]     = Bs[Nl * 32 + kk];
        bf.h[2 * v + 1] = Bs[Nl * 32 + kk + 1];
      }
#pragma unroll
      for (int g = 0; g < 2; ++g)
        acc[g][nt].v = __builtin_amdgcn_wmma_f32_16x16x32_f16(
            false, a[g].v, false, bf.v, (short)0, acc[g][nt].v, false, false);
    }
    __syncthreads();
  }

  float gateval = 1.0f;
  if constexpr (MODE == 1) gateval = 1.0f / (1.0f + expf(-gate[0]));

#pragma unroll
  for (int g = 0; g < 2; ++g)
#pragma unroll
    for (int nt = 0; nt < 4; ++nt) {
      int gcol = n0 + no + nt * 16 + Ml;
#pragma unroll
      for (int r = 0; r < 8; ++r) {
        int grow = m0 + mo + g * 16 + r + hi8;
        float val = acc[g][nt].f[r] + bias[gcol];
        if constexpr (GELU)
          val = 0.5f * val * (1.0f + erff(val * 0.70710678118654752f));
        size_t oi = (size_t)grow * N + gcol;
        if constexpr (MODE == 0) out[oi] = (OT)val;
        if constexpr (MODE == 1) out[oi] = out[oi] + gateval * val;
        if constexpr (MODE == 2) out[oi] = (OT)(resid[oi] + val);
      }
    }
}

// ---------------------------------------------------------------------------
// Stride-8 8x8 conv as WMMA GEMM. Rows = 64 kv tokens of one batch image,
// cols = c_out, K = c_in*64 + (kh*8+kw)  => weight reads are contiguous
// (cw[n][k] row-major), the token gather lands on the smaller A tile.
// Block 256 thr, tile 64(M) x 128(N), wave tile 16x64, K-step 32.
// ---------------------------------------------------------------------------
__global__ __launch_bounds__(256) void conv_gemm_kernel(
    const float* __restrict__ kvx, const float* __restrict__ cw,
    const float* __restrict__ cb, float* __restrict__ kvt) {
  __shared__ _Float16 As[64 * 32];
  __shared__ _Float16 Bs[128 * 32];

  const int tid  = threadIdx.x;
  const int lane = tid & 31, wv = tid >> 5;
  const int b = blockIdx.y, n0 = blockIdx.x * 128;
  const int mo = (wv >> 1) * 16, no = (wv & 1) * 64;
  const int Ml = lane & 15, half = lane >> 4;
  const int hi8 = half * 8, hi16 = half * 16;

  AccF acc[4];
#pragma unroll
  for (int nt = 0; nt < 4; ++nt)
#pragma unroll
    for (int e = 0; e < 8; ++e) acc[nt].f[e] = 0.0f;

  const int trow = tid >> 2, akc = (tid & 3) * 8;    // A: 64 rows x 32 kc
  const int ty = trow >> 3, tx = trow & 7;
  const int bn = tid & 127, bkc = (tid >> 7) * 16;   // B: 128 n x 32 kc
  const float4* bwp =
      reinterpret_cast<const float4*>(cw + (size_t)(n0 + bn) * (DIM_ * 64));

  for (int k0 = 0; k0 < DIM_ * 64; k0 += 32) {
    const int ci = k0 >> 6, p0 = k0 & 63;
    // ---- stage A (token gather), 8 contiguous kc per thread -> 1 b128 store
    {
      Pack8 pk;
#pragma unroll
      for (int i = 0; i < 8; ++i) {
        int p = p0 + akc + i;
        int tok = (ty * 8 + (p >> 3)) * 64 + (tx * 8 + (p & 7));
        pk.h[i] = (_Float16)kvx[((size_t)b * NTOK_ + tok) * DIM_ + ci];
      }
      *reinterpret_cast<uint4*>(&As[trow * 32 + akc]) = pk.u;
    }
    // ---- stage B: 16 contiguous weights per thread (vector loads)
    {
      const float4* bp = bwp + ((k0 + bkc) >> 2);
      Pack8 p0k, p1k;
#pragma unroll
      for (int u = 0; u < 2; ++u) {
        float4 t0 = bp[2 * u], t1 = bp[2 * u + 1];
        Pack8& pk = u ? p1k : p0k;
        pk.h[0] = (_Float16)t0.x; pk.h[1] = (_Float16)t0.y;
        pk.h[2] = (_Float16)t0.z; pk.h[3] = (_Float16)t0.w;
        pk.h[4] = (_Float16)t1.x; pk.h[5] = (_Float16)t1.y;
        pk.h[6] = (_Float16)t1.z; pk.h[7] = (_Float16)t1.w;
      }
      *reinterpret_cast<uint4*>(&Bs[bn * 32 + bkc])     = p0k.u;
      *reinterpret_cast<uint4*>(&Bs[bn * 32 + bkc + 8]) = p1k.u;
    }
    __syncthreads();

    FragH a;
#pragma unroll
    for (int v = 0; v < 8; ++v) {
      int k = ((v & 3) << 1) + ((v >> 2) << 4) + hi8;
      a.h[2 * v]     = As[(mo + Ml) * 32 + k];
      a.h[2 * v + 1] = As[(mo + Ml) * 32 + k + 1];
    }
#pragma unroll
    for (int nt = 0; nt < 4; ++nt) {
      FragH bf;
      int Nl = no + nt * 16 + Ml;
#pragma unroll
      for (int v = 0; v < 8; ++v) {
        int kk = ((v & 3) << 1) + ((v >> 2) << 4) + hi8;
        bf.h[2 * v]     = Bs[Nl * 32 + kk];
        bf.h[2 * v + 1] = Bs[Nl * 32 + kk + 1];
      }
      acc[nt].v = __builtin_amdgcn_wmma_f32_16x16x32_f16(
          false, a.v, false, bf.v, (short)0, acc[nt].v, false, false);
    }
    __syncthreads();
  }

#pragma unroll
  for (int nt = 0; nt < 4; ++nt) {
    int gcol = n0 + no + nt * 16 + Ml;
#pragma unroll
    for (int r = 0; r < 8; ++r) {
      int t = mo + r + hi8;
      kvt[(size_t)(b * MKV_ + t) * DIM_ + gcol] = acc[nt].f[r] + cb[gcol];
    }
  }
}

// ---------------------------------------------------------------------------
// LayerNorm over last dim (512), one wave32 per row.
// ---------------------------------------------------------------------------
template <typename OT>
__global__ __launch_bounds__(256) void ln_kernel(
    const float* __restrict__ x, const float* __restrict__ g,
    const float* __restrict__ bb, OT* __restrict__ y, int rows) {
  int lane = threadIdx.x & 31, wv = threadIdx.x >> 5;
  int row = blockIdx.x * 8 + wv;
  if (row >= rows) return;
  const float* xr = x + (size_t)row * DIM_;
  float s = 0.0f, s2 = 0.0f;
  for (int i = lane; i < DIM_; i += 32) {
    float v = xr[i];
    s += v;
    s2 += v * v;
  }
#pragma unroll
  for (int off = 16; off >= 1; off >>= 1) {
    s  += __shfl_xor(s, off, 32);
    s2 += __shfl_xor(s2, off, 32);
  }
  float mean = s * (1.0f / DIM_);
  float var  = s2 * (1.0f / DIM_) - mean * mean;
  float inv  = rsqrtf(var + 1e-5f);
  OT* yr = y + (size_t)row * DIM_;
  for (int i = lane; i < DIM_; i += 32)
    yr[i] = (OT)((xr[i] - mean) * inv * g[i] + bb[i]);
}

// ---------------------------------------------------------------------------
// Fused SRA attention for one (b, h, 128 q-rows): S = q k^T * scale + bias,
// softmax over m=64, O = P v. q is f16 [B*N, 512]; k,v f32 [B*64, 512].
// Each of 8 waves owns 16 q rows; k (8KB) and v^T (8KB) staged in LDS.
// ---------------------------------------------------------------------------
__global__ __launch_bounds__(256) void attention_kernel(
    const _Float16* __restrict__ q, const float* __restrict__ kbuf,
    const float* __restrict__ vbuf, const float* __restrict__ biasr,
    _Float16* __restrict__ aout) {
  __shared__ _Float16 kls[64 * 64];     // [m][d]  (B of S-gemm, stored [N][K])
  __shared__ _Float16 vls[64 * 64];     // [d][m]  (B of O-gemm, stored [N][K])
  __shared__ _Float16 pls[8][16 * 64];  // per-wave probability tile [row][m]

  const int tid  = threadIdx.x;
  const int lane = tid & 31, wv = tid >> 5;
  const int h = blockIdx.y, b = blockIdx.z;
  const int Ml = lane & 15, half = lane >> 4;
  const int hi8 = half * 8, hi16 = half * 16;
  const int nbase = blockIdx.x * 128 + wv * 16;

  // stage k (packed b128) and v^T (transpose scatter), f32 -> f16
  {
    int m = tid >> 2, d0 = (tid & 3) * 16;
    const float4* kp = reinterpret_cast<const float4*>(
        kbuf + ((size_t)(b * MKV_ + m)) * DIM_ + h * DH_ + d0);
    Pack8 p0k, p1k;
#pragma unroll
    for (int u = 0; u < 2; ++u) {
      float4 t0 = kp[2 * u], t1 = kp[2 * u + 1];
      Pack8& pk = u ? p1k : p0k;
      pk.h[0] = (_Float16)t0.x; pk.h[1] = (_Float16)t0.y;
      pk.h[2] = (_Float16)t0.z; pk.h[3] = (_Float16)t0.w;
      pk.h[4] = (_Float16)t1.x; pk.h[5] = (_Float16)t1.y;
      pk.h[6] = (_Float16)t1.z; pk.h[7] = (_Float16)t1.w;
    }
    *reinterpret_cast<uint4*>(&kls[m * 64 + d0])     = p0k.u;
    *reinterpret_cast<uint4*>(&kls[m * 64 + d0 + 8]) = p1k.u;

    const float4* vp = reinterpret_cast<const float4*>(
        vbuf + ((size_t)(b * MKV_ + m)) * DIM_ + h * DH_ + d0);
#pragma unroll
    for (int u = 0; u < 4; ++u) {
      float4 t = vp[u];
      vls[(d0 + 4 * u + 0) * 64 + m] = (_Float16)t.x;
      vls[(d0 + 4 * u + 1) * 64 + m] = (_Float16)t.y;
      vls[(d0 + 4 * u + 2) * 64 + m] = (_Float16)t.z;
      vls[(d0 + 4 * u + 3) * 64 + m] = (_Float16)t.w;
    }
  }
  __syncthreads();

  // q A-fragments for K = d = 0..63 (two k-steps of 32)
  const _Float16* qrow =
      q + ((size_t)(b * NTOK_ + nbase + Ml)) * DIM_ + h * DH_;
  FragH aq[2];
#pragma unroll
  for (int ks = 0; ks < 2; ++ks)
#pragma unroll
    for (int v = 0; v < 8; ++v) {
      int k = ((v & 3) << 1) + ((v >> 2) << 4) + hi8 + ks * 32;
      aq[ks].h[2 * v]     = qrow[k];
      aq[ks].h[2 * v + 1] = qrow[k + 1];
    }

  // S = q k^T
  AccF s[4];
#pragma unroll
  for (int nt = 0; nt < 4; ++nt)
#pragma unroll
    for (int e = 0; e < 8; ++e) s[nt].f[e] = 0.0f;
#pragma unroll
  for (int ks = 0; ks < 2; ++ks)
#pragma unroll
    for (int nt = 0; nt < 4; ++nt) {
      FragH bf;
      int Nl = nt * 16 + Ml;
#pragma unroll
      for (int v = 0; v < 8; ++v) {
        int kk = ((v & 3) << 1) + ((v >> 2) << 4) + hi8 + ks * 32;
        bf.h[2 * v]     = kls[Nl * 64 + kk];
        bf.h[2 * v + 1] = kls[Nl * 64 + kk + 1];
      }
      s[nt].v = __builtin_amdgcn_wmma_f32_16x16x32_f16(
          false, aq[ks].v, false, bf.v, (short)0, s[nt].v, false, false);
    }

  // softmax over m (columns), scale 1/8, + resized bias
  float bi[4];
#pragma unroll
  for (int nt = 0; nt < 4; ++nt) bi[nt] = biasr[b * MKV_ + nt * 16 + Ml];

#pragma unroll
  for (int r = 0; r < 8; ++r) {
    float xv[4];
#pragma unroll
    for (int nt = 0; nt < 4; ++nt) xv[nt] = s[nt].f[r] * 0.125f + bi[nt];
    float mx = fmaxf(fmaxf(xv[0], xv[1]), fmaxf(xv[2], xv[3]));
#pragma unroll
    for (int off = 8; off >= 1; off >>= 1)
      mx = fmaxf(mx, __shfl_xor(mx, off, 32));  // stays within 16-lane half
    float ex[4], sm = 0.0f;
#pragma unroll
    for (int nt = 0; nt < 4; ++nt) {
      ex[nt] = expf(xv[nt] - mx);
      sm += ex[nt];
    }
#pragma unroll
    for (int off = 8; off >= 1; off >>= 1) sm += __shfl_xor(sm, off, 32);
    float inv = 1.0f / sm;
    int row = r + 8 * half;
#pragma unroll
    for (int nt = 0; nt < 4; ++nt)
      pls[wv][row * 64 + nt * 16 + Ml] = (_Float16)(ex[nt] * inv);
  }
  __syncthreads();

  // P A-fragments (K = m = 0..63)
  FragH ap[2];
#pragma unroll
  for (int ks = 0; ks < 2; ++ks)
#pragma unroll
    for (int v = 0; v < 8; ++v) {
      int k = ((v & 3) << 1) + ((v >> 2) << 4) + hi8 + ks * 32;
      ap[ks].h[2 * v]     = pls[wv][Ml * 64 + k];
      ap[ks].h[2 * v + 1] = pls[wv][Ml * 64 + k + 1];
    }

  // O = P v
  AccF o4[4];
#pragma unroll
  for (int dt = 0; dt < 4; ++dt)
#pragma unroll
    for (int e = 0; e < 8; ++e) o4[dt].f[e] = 0.0f;
#pragma unroll
  for (int ks = 0; ks < 2; ++ks)
#pragma unroll
    for (int dt = 0; dt < 4; ++dt) {
      FragH bf;
      int Nl = dt * 16 + Ml;
#pragma unroll
      for (int v = 0; v < 8; ++v) {
        int kk = ((v & 3) << 1) + ((v >> 2) << 4) + hi8 + ks * 32;
        bf.h[2 * v]     = vls[Nl * 64 + kk];
        bf.h[2 * v + 1] = vls[Nl * 64 + kk + 1];
      }
      o4[dt].v = __builtin_amdgcn_wmma_f32_16x16x32_f16(
          false, ap[ks].v, false, bf.v, (short)0, o4[dt].v, false, false);
    }

#pragma unroll
  for (int dt = 0; dt < 4; ++dt) {
    int c = h * DH_ + dt * 16 + Ml;
#pragma unroll
    for (int r = 0; r < 8; ++r) {
      int n = nbase + r + 8 * half;
      aout[((size_t)(b * NTOK_ + n)) * DIM_ + c] = (_Float16)o4[dt].f[r];
    }
  }
}

// ---------------------------------------------------------------------------
// Host-side launch
// ---------------------------------------------------------------------------
extern "C" void kernel_launch(void* const* d_in, const int* in_sizes, int n_in,
                              void* d_out, int out_size, void* d_ws,
                              size_t ws_size, hipStream_t stream) {
  (void)in_sizes; (void)n_in; (void)out_size; (void)ws_size;

  // Input order: vf, if_, h, w, vis_bias, ir_bias, attn_bias_gamma,
  // params{v2i{qw,qb,kw,kb,vw,vb,projw,projb,convw,convb,ng,nb},
  //        i2v{...}, a, b, norm_g, norm_b, ffn_w1, ffn_b1, ffn_w2, ffn_b2}
  const float* vf    = (const float*)d_in[0];
  const float* if_   = (const float*)d_in[1];
  const float* visb  = (const float*)d_in[4];
  const float* irb   = (const float*)d_in[5];
  const float* gamma = (const float*)d_in[6];
  const int P0 = 7;   // v2i params base
  const int P1 = 19;  // i2v params base
  const float* gate_a = (const float*)d_in[31];
  const float* gate_b = (const float*)d_in[32];
  const float* ng     = (const float*)d_in[33];
  const float* nb     = (const float*)d_in[34];
  const float* fw1    = (const float*)d_in[35];
  const float* fb1    = (const float*)d_in[36];
  const float* fw2    = (const float*)d_in[37];
  const float* fb2    = (const float*)d_in[38];

  // workspace carve-up (256B aligned)
  char* ws = (char*)d_ws;
  size_t off = 0;
  auto carve = [&](size_t bytes) -> char* {
    size_t a = (off + 255) & ~(size_t)255;
    off = a + bytes;
    return ws + a;
  };
  const size_t ROWS = (size_t)BATCH_ * NTOK_;  // 32768
  float*    biasr = (float*)carve(BATCH_ * MKV_ * 4);
  float*    kvt   = (float*)carve((size_t)BATCH_ * MKV_ * DIM_ * 4);
  float*    kvn   = (float*)carve((size_t)BATCH_ * MKV_ * DIM_ * 4);
  float*    kbuf  = (float*)carve((size_t)BATCH_ * MKV_ * DIM_ * 4);
  float*    vbuf  = (float*)carve((size_t)BATCH_ * MKV_ * DIM_ * 4);
  _Float16* qf16  = (_Float16*)carve(ROWS * DIM_ * 2);
  _Float16* aof16 = (_Float16*)carve(ROWS * DIM_ * 2);
  float*    xbuf  = (float*)carve(ROWS * DIM_ * 4);
  _Float16* ybuf  = (_Float16*)carve(ROWS * DIM_ * 2);
  _Float16* hbuf  = (_Float16*)carve(ROWS * HID_ * 2);

  // x = vf (residual accumulator)
  hipMemcpyAsync(xbuf, vf, ROWS * DIM_ * 4, hipMemcpyDeviceToDevice, stream);

  for (int br = 0; br < 2; ++br) {
    const float* qx   = br ? if_ : vf;
    const float* kvx  = br ? vf : if_;
    const float* bin  = br ? visb : irb;
    const float* gate = br ? gate_b : gate_a;
    const int    PB   = br ? P1 : P0;
    const float* q_w    = (const float*)d_in[PB + 0];
    const float* q_b    = (const float*)d_in[PB + 1];
    const float* k_w    = (const float*)d_in[PB + 2];
    const float* k_b    = (const float*)d_in[PB + 3];
    const float* v_w    = (const float*)d_in[PB + 4];
    const float* v_b    = (const float*)d_in[PB + 5];
    const float* proj_w = (const float*)d_in[PB + 6];
    const float* proj_b = (const float*)d_in[PB + 7];
    const float* conv_w = (const float*)d_in[PB + 8];
    const float* conv_b = (const float*)d_in[PB + 9];
    const float* sra_g  = (const float*)d_in[PB + 10];
    const float* sra_b  = (const float*)d_in[PB + 11];

    bias_resize_kernel<<<2, 256, 0, stream>>>(bin, gamma, biasr);
    conv_gemm_kernel<<<dim3(4, BATCH_), 256, 0, stream>>>(kvx, conv_w,
                                                          conv_b, kvt);
    ln_kernel<float><<<64, 256, 0, stream>>>(kvt, sra_g, sra_b, kvn,
                                             BATCH_ * MKV_);
    // q projection -> f16
    gemm_kernel<float, _Float16, false, 0><<<dim3(4, 256), 256, 0, stream>>>(
        qx, q_w, q_b, nullptr, nullptr, qf16, (int)ROWS, DIM_, DIM_);
    // k, v projections -> f32
    gemm_kernel<float, float, false, 0><<<dim3(4, 4), 256, 0, stream>>>(
        kvn, k_w, k_b, nullptr, nullptr, kbuf, BATCH_ * MKV_, DIM_, DIM_);
    gemm_kernel<float, float, false, 0><<<dim3(4, 4), 256, 0, stream>>>(
        kvn, v_w, v_b, nullptr, nullptr, vbuf, BATCH_ * MKV_, DIM_, DIM_);
    // fused attention
    attention_kernel<<<dim3(32, HEADS_, BATCH_), 256, 0, stream>>>(
        qf16, kbuf, vbuf, biasr, aof16);
    // output projection, gated-accumulated into x
    gemm_kernel<_Float16, float, false, 1><<<dim3(4, 256), 256, 0, stream>>>(
        aof16, proj_w, proj_b, gate, nullptr, xbuf, (int)ROWS, DIM_, DIM_);
  }

  // y = LN(x) -> f16 ; h = gelu(y @ W1 + b1) ; out = x + h @ W2 + b2
  ln_kernel<_Float16><<<4096, 256, 0, stream>>>(xbuf, ng, nb, ybuf,
                                                (int)ROWS);
  gemm_kernel<_Float16, _Float16, true, 0><<<dim3(8, 256), 256, 0, stream>>>(
      ybuf, fw1, fb1, nullptr, nullptr, hbuf, (int)ROWS, HID_, DIM_);
  gemm_kernel<_Float16, float, false, 2><<<dim3(4, 256), 256, 0, stream>>>(
      hbuf, fw2, fb2, nullptr, xbuf, (float*)d_out, (int)ROWS, DIM_, HID_);
}
